// MultiheadAttentionMechanism_27161373180260
// MI455X (gfx1250) — matrix operand order repaired and checked by reading, not compiled, not run on hardware
//
#include <hip/hip_runtime.h>
#include <hip/hip_bf16.h>
#include <math.h>

typedef __attribute__((ext_vector_type(16))) _Float16 v16h;
typedef __attribute__((ext_vector_type(8)))  _Float16 v8h;
typedef __attribute__((ext_vector_type(8)))  float    v8f;

// Fast device transcendentals: gfx1250 has V_TANH_F32 (TRANS pipe).
#if __has_builtin(__builtin_amdgcn_tanhf)
__device__ __forceinline__ float fast_tanh(float x) { return __builtin_amdgcn_tanhf(x); }
#elif __has_builtin(__builtin_amdgcn_tanh_f32)
__device__ __forceinline__ float fast_tanh(float x) { return __builtin_amdgcn_tanh_f32(x); }
#else
__device__ __forceinline__ float fast_tanh(float x) { return tanhf(x); }
#endif

// Problem constants
#define BB   32
#define TT   1500
#define EE   512
#define AA   512
#define HH   4
#define CC   10
#define KK   201
#define HA   2048          // H*A = N of the big GEMM
#define KEXT 576           // 512 (enc) + 40 (conv feats) + 24 pad
#define KT   18            // KEXT / 32
#define NT   128           // HA / 16
#define MT_WG 128          // rows per workgroup in the fused GEMM
#define TBLK 12            // ceil(1500/128)

// Workspace layout (byte offsets, all 256B aligned)
#define WS_DECF 0u                               // B*HA f32            = 262144 B
#define WS_CONV 262144u                          // B*T*H*C f32         = 7680000 B
#define WS_BPK  7942144u                         // NT*KT*512 f16       = 2359296 B
#define WS_ENER 10301440u                        // B*T*H f32           = 768000 B
#define WS_CTX  11069440u                        // B*H*E f32           = 262144 B

// ---------------------------------------------------------------------------
// K1: pack w_enc (+ block-diagonal w_conv extension) into WMMA B-fragment
// layout, f32 -> f16.  Fragment (nt,kt): 32 lanes x 16 f16 contiguous; lane l
// holds N = nt*16 + (l&15), K = kt*32 + (l>>4)*16 + e  (e = 0..15).
// ---------------------------------------------------------------------------
__global__ __launch_bounds__(256) void pack_weights(
    const float* __restrict__ w_enc, const float* __restrict__ w_conv,
    _Float16* __restrict__ bpk)
{
  int tid  = blockIdx.x * 256 + threadIdx.x;     // 576*256 = 147456 threads
  int frag = tid >> 6;                           // 0..2303  (= nt*KT + kt)
  int within = tid & 63;
  int half = within >> 5;
  int lane = within & 31;
  int nt = frag / KT;
  int kt = frag % KT;
  int n = nt * 16 + (lane & 15);
  int h = n >> 9;
  int a = n & 511;
  int kbase = kt * 32 + ((lane >> 4) << 4) + (half << 3);
  _Float16 vals[8];
#pragma unroll
  for (int i = 0; i < 8; ++i) {
    int k = kbase + i;
    float v = 0.0f;
    if (k < EE) {
      v = w_enc[((size_t)h * EE + k) * AA + a];
    } else if (k < EE + HH * CC) {
      int hc = k - EE;                           // hc = h'*10 + c
      if ((hc / CC) == h) v = w_conv[(size_t)hc * AA + a];
    }
    vals[i] = (_Float16)v;
  }
  _Float16* dst = bpk + ((size_t)frag << 9) + (lane << 4) + (half << 3);
  *(v8h*)dst = *(const v8h*)vals;
}

// ---------------------------------------------------------------------------
// K2: dec_full[b, h*A+a] = b_enc[h,a] + sum_d dec_out[b,d] * w_dec[h,d,a]
// ---------------------------------------------------------------------------
__global__ __launch_bounds__(256) void dec_proj(
    const float* __restrict__ dec_out, const float* __restrict__ w_dec,
    const float* __restrict__ b_enc, float* __restrict__ decf)
{
  int b = blockIdx.y;
  int n = blockIdx.x * 256 + threadIdx.x;        // 0..2047
  int h = n >> 9;
  int a = n & 511;
  const float* dvec = dec_out + (size_t)b * EE;  // uniform per block -> s_loads
  const float* wcol = w_dec + (size_t)h * EE * AA + a;
  float acc = b_enc[n];
  for (int d = 0; d < EE; ++d)
    acc = fmaf(dvec[d], wcol[(size_t)d * AA], acc);
  decf[(size_t)b * HA + n] = acc;
}

// ---------------------------------------------------------------------------
// K3: location conv. conv_feat[b,t,h,c] = sum_k conv_k[h*C+c,k]*aw[b,t+k-100,h]
// Coefficients are wave-uniform (h from blockIdx, c/k loop constants) -> SMEM.
// ---------------------------------------------------------------------------
__global__ __launch_bounds__(256) void loc_conv(
    const float* __restrict__ aw_step, const float* __restrict__ conv_k,
    float* __restrict__ convf)
{
  __shared__ float awl[456];
  int tc = blockIdx.x, h = blockIdx.y, b = blockIdx.z;
  int tid = threadIdx.x;
  int w0 = tc * 256 - 100;
  for (int i = tid; i < 456; i += 256) {
    int t = w0 + i;
    awl[i] = (t >= 0 && t < TT) ? aw_step[((size_t)b * TT + t) * HH + h] : 0.0f;
  }
  __syncthreads();
  int t = tc * 256 + tid;
  if (t >= TT) return;
  float acc[CC];
#pragma unroll
  for (int c = 0; c < CC; ++c) acc[c] = 0.0f;
  const float* kc = conv_k + (size_t)h * CC * KK;   // uniform base
  for (int k = 0; k < KK; ++k) {
    float a = awl[tid + k];
#pragma unroll
    for (int c = 0; c < CC; ++c)
      acc[c] = fmaf(kc[c * KK + k], a, acc[c]);
  }
  float* dst = convf + (((size_t)b * TT + t) * HH + h) * CC;
#pragma unroll
  for (int c = 0; c < CC; ++c) dst[c] = acc[c];
}

// ---------------------------------------------------------------------------
// K4: fused [128 x 576] x [576 x 2048] f16 WMMA GEMM + energy epilogue.
// Wave w: head h = w>>1, rows mtg = w&1 (4 M-tiles of 16). Each wave covers
// the head's full A-range (32 N-tiles) -> exclusive energy writes.
// A tile staged in LDS pre-swizzled to WMMA A-fragment layout
//   (lane l: M = l&15, K = kt*32 + (l>>4)*16 + e, e = 0..15).
// ---------------------------------------------------------------------------
__global__ __launch_bounds__(256) void fused_gemm_energy(
    const float* __restrict__ enc_out, const float* __restrict__ convf,
    const _Float16* __restrict__ bpk, const float* __restrict__ decf,
    const float* __restrict__ vvec, float* __restrict__ energy)
{
  extern __shared__ _Float16 As[];               // 8*18*32*16 f16 = 144 KB
  const int tid = threadIdx.x;
  const int b = blockIdx.y;
  const int t0 = blockIdx.x * MT_WG;
  const float* encb = enc_out + (size_t)b * TT * EE;
  const float* cfb  = convf   + (size_t)b * TT * (HH * CC);

  // ---- stage A tile (swizzled, f32 -> f16, zero-pad rows/cols) ----
  for (int i = tid; i < 8 * KT * 32 * 16; i += 256) {
    int pos  = i & 15;
    int lane = (i >> 4) & 31;
    int r3   = i >> 9;
    int kt   = r3 % KT;
    int mt   = r3 / KT;
    int m = mt * 16 + (lane & 15);
    int g = pos >> 3;
    int k = kt * 32 + ((lane >> 4) << 4) + pos;
    int t = t0 + m;
    float v = 0.0f;
    if (t < TT) {
      if (k < EE) v = encb[(size_t)t * EE + k];
      else if (k < EE + HH * CC) v = cfb[(size_t)t * (HH * CC) + (k - EE)];
    }
    As[i] = (_Float16)v;
  }
  __syncthreads();

  const int w = tid >> 5, lane = tid & 31;
  const int h = w >> 1, mtg = w & 1;
  const int lm = lane & 15, lh = lane >> 4;

  float part[4][8];
#pragma unroll
  for (int mi = 0; mi < 4; ++mi)
#pragma unroll
    for (int j = 0; j < 8; ++j) part[mi][j] = 0.0f;

  for (int ng = 0; ng < 8; ++ng) {               // 4 N-tiles per group
    v8f acc[4][4];
#pragma unroll
    for (int mi = 0; mi < 4; ++mi)
#pragma unroll
      for (int q = 0; q < 4; ++q)
#pragma unroll
        for (int j = 0; j < 8; ++j) acc[mi][q][j] = 0.0f;

    int ntb = h * 32 + ng * 4;
    for (int kt = 0; kt < KT; ++kt) {
      v16h af[4];
#pragma unroll
      for (int mi = 0; mi < 4; ++mi) {
        int mt = mtg * 4 + mi;
        af[mi] = *(const v16h*)(As + (((size_t)(mt * KT + kt) * 32 + lane) << 4));
      }
#pragma unroll
      for (int q = 0; q < 4; ++q) {
        size_t frag = (size_t)(ntb + q) * KT + kt;
        v16h bf = *(const v16h*)(bpk + (frag << 9) + (lane << 4));
#pragma unroll
        for (int mi = 0; mi < 4; ++mi)
          acc[mi][q] = __builtin_amdgcn_wmma_f32_16x16x32_f16(
              false, af[mi], false, bf, (short)0, acc[mi][q], false, false);
      }
    }
    // epilogue: + dec_a, tanh (HW TRANS op), * v, partial reduce over N
#pragma unroll
    for (int q = 0; q < 4; ++q) {
      int n = (ntb + q) * 16 + lm;
      float dec = decf[(size_t)b * HA + n];
      float vv  = vvec[n];
#pragma unroll
      for (int mi = 0; mi < 4; ++mi)
#pragma unroll
        for (int j = 0; j < 8; ++j)
          part[mi][j] += vv * fast_tanh(acc[mi][q][j] + dec);
    }
  }

  // reduce across the 16 lanes sharing each row (xor within 16-lane halves)
#pragma unroll
  for (int mi = 0; mi < 4; ++mi)
#pragma unroll
    for (int j = 0; j < 8; ++j) {
      float s = part[mi][j];
      s += __shfl_xor(s, 1);
      s += __shfl_xor(s, 2);
      s += __shfl_xor(s, 4);
      s += __shfl_xor(s, 8);
      part[mi][j] = s;
    }
  if (lm == 0) {
#pragma unroll
    for (int mi = 0; mi < 4; ++mi)
#pragma unroll
      for (int j = 0; j < 8; ++j) {
        int row = (mtg * 4 + mi) * 16 + lh * 8 + j;
        int t = t0 + row;
        if (t < TT) energy[((size_t)b * TT + t) * HH + h] = part[mi][j];
      }
  }
}

// ---------------------------------------------------------------------------
// K5: masked softmax over T per (b,h):  e' = e * (t<len ? 1 : -1024)
// ---------------------------------------------------------------------------
__global__ __launch_bounds__(256) void softmax_t(
    const float* __restrict__ energy, const int* __restrict__ x_lens,
    float* __restrict__ awout)
{
  __shared__ float red[8];
  int b = blockIdx.x >> 2, h = blockIdx.x & 3;
  int tid = threadIdx.x;
  int len = x_lens[b];
  float ev[6];
#pragma unroll
  for (int i = 0; i < 6; ++i) {
    int t = tid + i * 256;
    float e = -INFINITY;
    if (t < TT) {
      float x = energy[((size_t)b * TT + t) * HH + h];
      e = x * ((t < len) ? 1.0f : -1024.0f);
    }
    ev[i] = e;
  }
  float mx = ev[0];
#pragma unroll
  for (int i = 1; i < 6; ++i) mx = fmaxf(mx, ev[i]);
#pragma unroll
  for (int m = 16; m >= 1; m >>= 1) mx = fmaxf(mx, __shfl_xor(mx, m));
  if ((tid & 31) == 0) red[tid >> 5] = mx;
  __syncthreads();
  mx = red[0];
#pragma unroll
  for (int r = 1; r < 8; ++r) mx = fmaxf(mx, red[r]);
  __syncthreads();

  float sm = 0.0f;
#pragma unroll
  for (int i = 0; i < 6; ++i) {
    int t = tid + i * 256;
    float ex = (t < TT) ? __expf(ev[i] - mx) : 0.0f;   // v_exp_f32 path
    ev[i] = ex;
    sm += ex;
  }
#pragma unroll
  for (int m = 16; m >= 1; m >>= 1) sm += __shfl_xor(sm, m);
  if ((tid & 31) == 0) red[tid >> 5] = sm;
  __syncthreads();
  sm = 0.0f;
#pragma unroll
  for (int r = 0; r < 8; ++r) sm += red[r];
  float inv = 1.0f / sm;
#pragma unroll
  for (int i = 0; i < 6; ++i) {
    int t = tid + i * 256;
    if (t < TT) awout[((size_t)b * TT + t) * HH + h] = ev[i] * inv;
  }
}

// ---------------------------------------------------------------------------
// K6: ctx_raw[b,h,e] = sum_t aw[b,t,h] * enc_out[b,t,e]  (streams enc once)
// ---------------------------------------------------------------------------
__global__ __launch_bounds__(256) void ctx_gemv(
    const float* __restrict__ enc_out, const float* __restrict__ awout,
    float* __restrict__ ctxraw)
{
  int b = blockIdx.y;
  int e = blockIdx.x * 256 + threadIdx.x;
  const float* eb = enc_out + (size_t)b * TT * EE;
  const float* aw = awout   + (size_t)b * TT * HH;   // uniform -> s_loads
  float acc[HH];
#pragma unroll
  for (int h = 0; h < HH; ++h) acc[h] = 0.0f;
  for (int t = 0; t < TT; ++t) {
    float x = eb[(size_t)t * EE + e];
#pragma unroll
    for (int h = 0; h < HH; ++h)
      acc[h] = fmaf(aw[t * HH + h], x, acc[h]);
  }
#pragma unroll
  for (int h = 0; h < HH; ++h)
    ctxraw[(size_t)b * HA + h * EE + e] = acc[h];
}

// ---------------------------------------------------------------------------
// K7: out[b,e] = b_mha[e] + sum_he ctx_raw[b,he] * w_mha[he,e]
// ---------------------------------------------------------------------------
__global__ __launch_bounds__(256) void mha_out(
    const float* __restrict__ ctxraw, const float* __restrict__ w_mha,
    const float* __restrict__ b_mha, float* __restrict__ out)
{
  int b = blockIdx.y;
  int e = blockIdx.x * 256 + threadIdx.x;
  const float* cr = ctxraw + (size_t)b * HA;         // uniform -> s_loads
  float acc = b_mha[e];
  for (int he = 0; he < HA; ++he)
    acc = fmaf(cr[he], w_mha[(size_t)he * EE + e], acc);
  out[(size_t)b * EE + e] = acc;
}

// ---------------------------------------------------------------------------
extern "C" void kernel_launch(void* const* d_in, const int* in_sizes, int n_in,
                              void* d_out, int out_size, void* d_ws, size_t ws_size,
                              hipStream_t stream) {
  (void)in_sizes; (void)n_in; (void)out_size; (void)ws_size;
  const float* enc_out = (const float*)d_in[0];
  const int*   x_lens  = (const int*)d_in[1];
  const float* dec_out = (const float*)d_in[2];
  const float* aw_step = (const float*)d_in[3];
  const float* w_enc   = (const float*)d_in[4];
  const float* b_enc   = (const float*)d_in[5];
  const float* w_dec   = (const float*)d_in[6];
  const float* w_conv  = (const float*)d_in[7];
  const float* conv_k  = (const float*)d_in[8];
  const float* vvec    = (const float*)d_in[9];
  const float* w_mha   = (const float*)d_in[10];
  const float* b_mha   = (const float*)d_in[11];

  char* ws = (char*)d_ws;
  float*    decf   = (float*)(ws + WS_DECF);
  float*    convf  = (float*)(ws + WS_CONV);
  _Float16* bpk    = (_Float16*)(ws + WS_BPK);
  float*    energy = (float*)(ws + WS_ENER);
  float*    ctxraw = (float*)(ws + WS_CTX);

  float* out     = (float*)d_out;
  float* ctx_out = out;                   // [B,1,E]
  float* aw_out  = out + BB * EE;         // [B,T,H]

  (void)hipFuncSetAttribute((const void*)fused_gemm_energy,
                            hipFuncAttributeMaxDynamicSharedMemorySize,
                            8 * KT * 32 * 16 * (int)sizeof(_Float16));

  pack_weights<<<576, 256, 0, stream>>>(w_enc, w_conv, bpk);
  dec_proj<<<dim3(8, BB), 256, 0, stream>>>(dec_out, w_dec, b_enc, decf);
  loc_conv<<<dim3(6, HH, BB), 256, 0, stream>>>(aw_step, conv_k, convf);
  fused_gemm_energy<<<dim3(TBLK, BB), 256,
                      8 * KT * 32 * 16 * sizeof(_Float16), stream>>>(
      enc_out, convf, bpk, decf, vvec, energy);
  softmax_t<<<BB * HH, 256, 0, stream>>>(energy, x_lens, aw_out);
  ctx_gemv<<<dim3(2, BB), 256, 0, stream>>>(enc_out, aw_out, ctxraw);
  mha_out<<<dim3(2, BB), 256, 0, stream>>>(ctxraw, w_mha, b_mha, ctx_out);
}
